// TransformerEncoderLayerWithMoE_76433237999750
// MI455X (gfx1250) — matrix operand order repaired and checked by reading, not compiled, not run on hardware
//
#include <hip/hip_runtime.h>

// ---------------- problem constants ----------------
static constexpr int D_MODEL = 1024;
static constexpr int NHEAD   = 16;
static constexpr int HEAD_DIM= 64;
static constexpr int D_HID   = 4096;
static constexpr int NEXP    = 8;
static constexpr int BATCH   = 4;
static constexpr int SEQ     = 2048;
static constexpr int TOK     = BATCH * SEQ;          // 8192
static constexpr float LN_EPS = 1e-5f;

// ---------------- vector types ----------------
typedef __attribute__((ext_vector_type(8)))  float  v8f;
typedef __attribute__((ext_vector_type(8)))  __bf16 v8bf;
typedef __attribute__((ext_vector_type(16))) __bf16 v16bf;

// ---------------- helpers ----------------
__device__ __forceinline__ __bf16 f2bf(float f) {
    unsigned u = __builtin_bit_cast(unsigned, f);
    unsigned r = u + 0x7FFFu + ((u >> 16) & 1u);   // round-to-nearest-even
    unsigned short h = (unsigned short)(r >> 16);
    return __builtin_bit_cast(__bf16, h);
}

__device__ __forceinline__ v16bf cat16(v8bf lo, v8bf hi) {
    v16bf r;
#pragma unroll
    for (int i = 0; i < 8; ++i) { r[i] = lo[i]; r[i + 8] = hi[i]; }
    return r;
}

// A fragment (16x32 bf16, MxK): lane m = lane&15; lanes 0-15 hold K 0-7,16-23;
// lanes 16-31 hold K 8-15,24-31.  Row pointer is per-lane (enables gather).
__device__ __forceinline__ v16bf load_a_row(const __bf16* row, int k0, int lane) {
    const int kh = ((lane >> 4) << 3);
    v8bf lo = *(const v8bf*)(row + k0 + kh);
    v8bf hi = *(const v8bf*)(row + k0 + 16 + kh);
    return cat16(lo, hi);
}

// B fragment (32x16 bf16, KxN) read from row-major [N,K] storage (x @ W^T):
// lane n = nbase + (lane&15); lanes 0-15 hold K 0-7,16-23, lanes 16-31 hold K 8-15,24-31.
__device__ __forceinline__ v16bf load_b_nt(const __bf16* B, long ldb, int nbase,
                                           int k0, int lane) {
    const __bf16* p = B + (long)(nbase + (lane & 15)) * ldb + k0 + ((lane >> 4) << 3);
    v8bf lo = *(const v8bf*)p;
    v8bf hi = *(const v8bf*)(p + 16);
    return cat16(lo, hi);
}

__device__ __forceinline__ v8f wmma_bf16(v16bf a, v16bf b, v8f c) {
    return __builtin_amdgcn_wmma_f32_16x16x32_bf16(false, a, false, b,
                                                   (short)0, c, false, false);
}

// ---------------- fp32 -> bf16 convert ----------------
__global__ void k_cvt(const float* __restrict__ in, __bf16* __restrict__ out, long n) {
    long i = (long)blockIdx.x * blockDim.x + threadIdx.x;
    if (i < n) out[i] = f2bf(in[i]);
}

// ---------------- QKV projection (32x64 wave tile) ----------------
// C[m,n] = src[m,:] . in_proj_w[n,:] + b[n];  M=8192, N=3072, K=1024.
// Writes q,k row-major [B,H,S,64] and v transposed [B,H,64,S] in bf16.
__global__ void __launch_bounds__(32)
k_qkv(const __bf16* __restrict__ X, const __bf16* __restrict__ W,
      const float* __restrict__ bias,
      __bf16* __restrict__ qb, __bf16* __restrict__ kb, __bf16* __restrict__ vt) {
    const int lane = threadIdx.x;
    const int nb = blockIdx.x * 64;
    const int mb = blockIdx.y * 32;
    const __bf16* arow0 = X + (long)(mb + (lane & 15)) * D_MODEL;
    const __bf16* arow1 = arow0 + (long)16 * D_MODEL;

    v8f acc[2][4];
#pragma unroll
    for (int h = 0; h < 2; ++h)
#pragma unroll
        for (int t = 0; t < 4; ++t) acc[h][t] = {};

    for (int k0 = 0; k0 < D_MODEL; k0 += 32) {
        v16bf a0 = load_a_row(arow0, k0, lane);
        v16bf a1 = load_a_row(arow1, k0, lane);
#pragma unroll
        for (int t = 0; t < 4; ++t) {
            v16bf b = load_b_nt(W, D_MODEL, nb + t * 16, k0, lane);
            acc[0][t] = wmma_bf16(a0, b, acc[0][t]);
            acc[1][t] = wmma_bf16(a1, b, acc[1][t]);
        }
    }
#pragma unroll
    for (int t = 0; t < 4; ++t) {
        const int n = nb + t * 16 + (lane & 15);
        const int part = n >> 10, rem = n & 1023;
        const int hh = rem >> 6, hd = rem & 63;
        const float bv = bias[n];
#pragma unroll
        for (int half = 0; half < 2; ++half) {
#pragma unroll
            for (int r = 0; r < 8; ++r) {
                const int m = mb + half * 16 + r + ((lane >> 4) << 3);
                const int bi = m >> 11, s = m & 2047;
                const __bf16 o = f2bf(acc[half][t][r] + bv);
                if (part == 0)
                    qb[(((long)bi * NHEAD + hh) * SEQ + s) * HEAD_DIM + hd] = o;
                else if (part == 1)
                    kb[(((long)bi * NHEAD + hh) * SEQ + s) * HEAD_DIM + hd] = o;
                else
                    vt[(((long)bi * NHEAD + hh) * HEAD_DIM + hd) * SEQ + s] = o;
            }
        }
    }
}

// ---------------- flash attention (one wave = 16 query rows) ----------------
__global__ void __launch_bounds__(32)
k_attn(const __bf16* __restrict__ qb, const __bf16* __restrict__ kb,
       const __bf16* __restrict__ vt, __bf16* __restrict__ ctx) {
    const int lane = threadIdx.x;
    const int bh = blockIdx.y;               // b*16 + h
    const int mb = blockIdx.x * 16;
    const __bf16* qbase = qb + (long)bh * SEQ * HEAD_DIM;
    const __bf16* kbase = kb + (long)bh * SEQ * HEAD_DIM;
    const __bf16* vbase = vt + (long)bh * HEAD_DIM * SEQ;
    const __bf16* qrow = qbase + (long)(mb + (lane & 15)) * HEAD_DIM;

    const v16bf a0 = load_a_row(qrow, 0, lane);
    const v16bf a1 = load_a_row(qrow, 32, lane);

    v8f o0 = {}, o1 = {}, o2 = {}, o3 = {};
    float mrun[8], lrun[8];
#pragma unroll
    for (int r = 0; r < 8; ++r) { mrun[r] = -1e30f; lrun[r] = 0.f; }

    __shared__ __attribute__((aligned(16))) __bf16 pbuf[16 * 32];
    const float scale = 0.125f;               // 1/sqrt(64)

    for (int s0 = 0; s0 < SEQ; s0 += 32) {
        v8f st0 = {}, st1 = {};
        st0 = wmma_bf16(a0, load_b_nt(kbase, HEAD_DIM, s0,      0,  lane), st0);
        st0 = wmma_bf16(a1, load_b_nt(kbase, HEAD_DIM, s0,      32, lane), st0);
        st1 = wmma_bf16(a0, load_b_nt(kbase, HEAD_DIM, s0 + 16, 0,  lane), st1);
        st1 = wmma_bf16(a1, load_b_nt(kbase, HEAD_DIM, s0 + 16, 32, lane), st1);

#pragma unroll
        for (int r = 0; r < 8; ++r) {
            float x0 = st0[r] * scale, x1 = st1[r] * scale;
            float mx = fmaxf(x0, x1);
#pragma unroll
            for (int w = 1; w < 16; w <<= 1) mx = fmaxf(mx, __shfl_xor(mx, w));
            const float newm = fmaxf(mrun[r], mx);
            const float corr = __expf(mrun[r] - newm);
            mrun[r] = newm;
            x0 = __expf(x0 - newm); x1 = __expf(x1 - newm);
            float rs = x0 + x1;
#pragma unroll
            for (int w = 1; w < 16; w <<= 1) rs += __shfl_xor(rs, w);
            lrun[r] = lrun[r] * corr + rs;
            o0[r] *= corr; o1[r] *= corr; o2[r] *= corr; o3[r] *= corr;
            const int m = r + ((lane >> 4) << 3);
            const int nn = lane & 15;
            pbuf[m * 32 + nn]      = f2bf(x0);
            pbuf[m * 32 + 16 + nn] = f2bf(x1);
        }
        __syncthreads();
        {   // C-layout -> A-layout via LDS, then P @ V
            const int kh = ((lane >> 4) << 3);
            const __bf16* prow = &pbuf[(lane & 15) * 32];
            v8bf lo = *(const v8bf*)(prow + kh);
            v8bf hi = *(const v8bf*)(prow + 16 + kh);
            v16bf pa = cat16(lo, hi);
            o0 = wmma_bf16(pa, load_b_nt(vbase, SEQ, 0,  s0, lane), o0);
            o1 = wmma_bf16(pa, load_b_nt(vbase, SEQ, 16, s0, lane), o1);
            o2 = wmma_bf16(pa, load_b_nt(vbase, SEQ, 32, s0, lane), o2);
            o3 = wmma_bf16(pa, load_b_nt(vbase, SEQ, 48, s0, lane), o3);
        }
        __syncthreads();
    }
    const int bi = bh >> 4, h = bh & 15;
    const int nn = lane & 15;
#pragma unroll
    for (int r = 0; r < 8; ++r) {
        const int m = mb + r + ((lane >> 4) << 3);
        const long t = (long)bi * SEQ + m;
        const float inv = 1.0f / lrun[r];
        __bf16* dst = ctx + t * D_MODEL + h * HEAD_DIM;
        dst[nn]      = f2bf(o0[r] * inv);
        dst[16 + nn] = f2bf(o1[r] * inv);
        dst[32 + nn] = f2bf(o2[r] * inv);
        dst[48 + nn] = f2bf(o3[r] * inv);
    }
}

// ---------------- out-proj + residual (32x64 wave tile) ----------------
__global__ void __launch_bounds__(32)
k_outproj(const __bf16* __restrict__ Ctx, const __bf16* __restrict__ W,
          const float* __restrict__ bias, const float* __restrict__ src,
          float* __restrict__ resid) {
    const int lane = threadIdx.x;
    const int nb = blockIdx.x * 64;
    const int mb = blockIdx.y * 32;
    const __bf16* arow0 = Ctx + (long)(mb + (lane & 15)) * D_MODEL;
    const __bf16* arow1 = arow0 + (long)16 * D_MODEL;

    v8f acc[2][4];
#pragma unroll
    for (int h = 0; h < 2; ++h)
#pragma unroll
        for (int t = 0; t < 4; ++t) acc[h][t] = {};

    for (int k0 = 0; k0 < D_MODEL; k0 += 32) {
        v16bf a0 = load_a_row(arow0, k0, lane);
        v16bf a1 = load_a_row(arow1, k0, lane);
#pragma unroll
        for (int t = 0; t < 4; ++t) {
            v16bf b = load_b_nt(W, D_MODEL, nb + t * 16, k0, lane);
            acc[0][t] = wmma_bf16(a0, b, acc[0][t]);
            acc[1][t] = wmma_bf16(a1, b, acc[1][t]);
        }
    }
#pragma unroll
    for (int t = 0; t < 4; ++t) {
        const int n = nb + t * 16 + (lane & 15);
        const float bv = bias[n];
#pragma unroll
        for (int half = 0; half < 2; ++half) {
#pragma unroll
            for (int r = 0; r < 8; ++r) {
                const int m = mb + half * 16 + r + ((lane >> 4) << 3);
                resid[(long)m * D_MODEL + n] =
                    acc[half][t][r] + bv + src[(long)m * D_MODEL + n];
            }
        }
    }
}

// ---------------- layernorm (optional add, fp32 + optional bf16 out) ----------------
__global__ void __launch_bounds__(256)
k_ln(const float* __restrict__ A, const float* __restrict__ Add,
     const float* __restrict__ g, const float* __restrict__ b,
     float* __restrict__ outf, __bf16* __restrict__ outbf) {
    const int row = blockIdx.x, tid = threadIdx.x;
    const float* a = A + (long)row * D_MODEL;
    float v[4]; float sum = 0.f;
#pragma unroll
    for (int i = 0; i < 4; ++i) {
        const int c = tid + i * 256;
        float x = a[c];
        if (Add) x += Add[(long)row * D_MODEL + c];
        v[i] = x; sum += x;
    }
    __shared__ float red[256];
    red[tid] = sum; __syncthreads();
    for (int s = 128; s > 0; s >>= 1) { if (tid < s) red[tid] += red[tid + s]; __syncthreads(); }
    const float mean = red[0] * (1.f / D_MODEL);
    __syncthreads();
    float vs = 0.f;
#pragma unroll
    for (int i = 0; i < 4; ++i) { const float d = v[i] - mean; vs += d * d; }
    red[tid] = vs; __syncthreads();
    for (int s = 128; s > 0; s >>= 1) { if (tid < s) red[tid] += red[tid + s]; __syncthreads(); }
    const float rstd = rsqrtf(red[0] * (1.f / D_MODEL) + LN_EPS);
#pragma unroll
    for (int i = 0; i < 4; ++i) {
        const int c = tid + i * 256;
        const float y = (v[i] - mean) * rstd * g[c] + b[c];
        if (outf)  outf[(long)row * D_MODEL + c] = y;
        if (outbf) outbf[(long)row * D_MODEL + c] = f2bf(y);
    }
}

// ---------------- gate: logits = x @ gate_w^T + gb  (tiny) ----------------
__global__ void __launch_bounds__(256)
k_gate(const float* __restrict__ xf, const float* __restrict__ gw,
       const float* __restrict__ gb, float* __restrict__ logits) {
    const int id = blockIdx.x * 256 + threadIdx.x;   // TOK*NEXP threads
    const int t = id >> 3, e = id & 7;
    const float* xr = xf + (long)t * D_MODEL;
    const float* wr = gw + (long)e * D_MODEL;
    float acc = 0.f;
    for (int k = 0; k < D_MODEL; ++k) acc += xr[k] * wr[k];
    logits[id] = acc + gb[e];
}

// ---------------- routing ----------------
__global__ void k_zero8(int* p) { if (threadIdx.x < 8 && blockIdx.x == 0) p[threadIdx.x] = 0; }

__global__ void __launch_bounds__(256)
k_route1(const float* __restrict__ logits, int* counts,
         int* tok_e, int* tok_pos, float* tok_p) {
    const int t = blockIdx.x * 256 + threadIdx.x;
    if (t >= TOK) return;
    float l[8];
#pragma unroll
    for (int e = 0; e < 8; ++e) l[e] = logits[t * 8 + e];
    int i0 = 0; float v0 = l[0];
#pragma unroll
    for (int e = 1; e < 8; ++e) if (l[e] > v0) { v0 = l[e]; i0 = e; }
    int i1 = -1; float v1 = -1e30f;
#pragma unroll
    for (int e = 0; e < 8; ++e) if (e != i0 && l[e] > v1) { v1 = l[e]; i1 = e; }
    const float e0 = 1.0f;                 // exp(v0 - v0)
    const float e1 = __expf(v1 - v0);
    const float inv = 1.0f / (e0 + e1);
    tok_e[2 * t] = i0;  tok_e[2 * t + 1] = i1;
    tok_p[2 * t] = e0 * inv;  tok_p[2 * t + 1] = e1 * inv;
    tok_pos[2 * t]     = atomicAdd(&counts[i0], 1);
    tok_pos[2 * t + 1] = atomicAdd(&counts[i1], 1);
}

__global__ void k_prefix(const int* counts, int* offs) {
    if (threadIdx.x == 0 && blockIdx.x == 0) {
        int o = 0;
        for (int e = 0; e < 8; ++e) { offs[e] = o; o += counts[e]; }
    }
}

__global__ void __launch_bounds__(256)
k_route2(const int* offs, const int* tok_e, const int* tok_pos,
         int* token_of, int* tok_slot) {
    const int i = blockIdx.x * 256 + threadIdx.x;
    if (i >= 2 * TOK) return;
    const int e = tok_e[i];
    const int slot = offs[e] + tok_pos[i];
    token_of[slot] = i >> 1;
    tok_slot[i] = slot;
}

// ---------------- expert layer 1: h = relu(x[gather] @ w1[e]^T + b1[e]) ----------------
// 32x64 wave tile with gathered A rows.
__global__ void __launch_bounds__(32)
k_ffn1(const __bf16* __restrict__ Xbf, const __bf16* __restrict__ W1,
       const float* __restrict__ b1, const int* __restrict__ counts,
       const int* __restrict__ offs, const int* __restrict__ token_of,
       __bf16* __restrict__ hbuf, int e) {
    const int cnt = counts[e];
    const int mb = blockIdx.y * 32;
    if (mb >= cnt) return;
    const int off = offs[e];
    const int lane = threadIdx.x;
    int slot0 = mb + (lane & 15);        if (slot0 >= cnt) slot0 = cnt - 1;
    int slot1 = mb + 16 + (lane & 15);   if (slot1 >= cnt) slot1 = cnt - 1;
    const __bf16* arow0 = Xbf + (long)token_of[off + slot0] * D_MODEL;
    const __bf16* arow1 = Xbf + (long)token_of[off + slot1] * D_MODEL;
    const __bf16* W = W1 + (long)e * D_HID * D_MODEL;
    const int nb = blockIdx.x * 64;

    v8f acc[2][4];
#pragma unroll
    for (int h = 0; h < 2; ++h)
#pragma unroll
        for (int t = 0; t < 4; ++t) acc[h][t] = {};

    for (int k0 = 0; k0 < D_MODEL; k0 += 32) {
        v16bf a0 = load_a_row(arow0, k0, lane);
        v16bf a1 = load_a_row(arow1, k0, lane);
#pragma unroll
        for (int t = 0; t < 4; ++t) {
            v16bf b = load_b_nt(W, D_MODEL, nb + t * 16, k0, lane);
            acc[0][t] = wmma_bf16(a0, b, acc[0][t]);
            acc[1][t] = wmma_bf16(a1, b, acc[1][t]);
        }
    }
#pragma unroll
    for (int t = 0; t < 4; ++t) {
        const int n = nb + t * 16 + (lane & 15);
        const float bv = b1[e * D_HID + n];
#pragma unroll
        for (int half = 0; half < 2; ++half) {
#pragma unroll
            for (int r = 0; r < 8; ++r) {
                const int sr = mb + half * 16 + r + ((lane >> 4) << 3);
                if (sr < cnt) {
                    float v = acc[half][t][r] + bv;
                    v = v > 0.f ? v : 0.f;
                    hbuf[(long)sr * D_HID + n] = f2bf(v);
                }
            }
        }
    }
}

// ---------------- expert layer 2: y[slot] = h[slot] @ w2[e]^T + b2[e] ----------------
__global__ void __launch_bounds__(32)
k_ffn2(const __bf16* __restrict__ hbuf, const __bf16* __restrict__ W2,
       const float* __restrict__ b2, const int* __restrict__ counts,
       const int* __restrict__ offs, float* __restrict__ ybuf, int e) {
    const int cnt = counts[e];
    const int mb = blockIdx.y * 32;
    if (mb >= cnt) return;
    const int off = offs[e];
    const int lane = threadIdx.x;
    int slot0 = mb + (lane & 15);        if (slot0 >= cnt) slot0 = cnt - 1;
    int slot1 = mb + 16 + (lane & 15);   if (slot1 >= cnt) slot1 = cnt - 1;
    const __bf16* arow0 = hbuf + (long)slot0 * D_HID;
    const __bf16* arow1 = hbuf + (long)slot1 * D_HID;
    const __bf16* W = W2 + (long)e * D_MODEL * D_HID;
    const int nb = blockIdx.x * 64;

    v8f acc[2][4];
#pragma unroll
    for (int h = 0; h < 2; ++h)
#pragma unroll
        for (int t = 0; t < 4; ++t) acc[h][t] = {};

    for (int k0 = 0; k0 < D_HID; k0 += 32) {
        v16bf a0 = load_a_row(arow0, k0, lane);
        v16bf a1 = load_a_row(arow1, k0, lane);
#pragma unroll
        for (int t = 0; t < 4; ++t) {
            v16bf b = load_b_nt(W, D_HID, nb + t * 16, k0, lane);
            acc[0][t] = wmma_bf16(a0, b, acc[0][t]);
            acc[1][t] = wmma_bf16(a1, b, acc[1][t]);
        }
    }
#pragma unroll
    for (int t = 0; t < 4; ++t) {
        const int n = nb + t * 16 + (lane & 15);
        const float bv = b2[e * D_MODEL + n];
#pragma unroll
        for (int half = 0; half < 2; ++half) {
#pragma unroll
            for (int r = 0; r < 8; ++r) {
                const int sr = mb + half * 16 + r + ((lane >> 4) << 3);
                if (sr < cnt)
                    ybuf[(long)(off + sr) * D_MODEL + n] = acc[half][t][r] + bv;
            }
        }
    }
}

// ---------------- top-2 combine: moe[t] = p0*y[slot0] + p1*y[slot1] ----------------
__global__ void __launch_bounds__(256)
k_moe(const float* __restrict__ y, const int* __restrict__ tok_slot,
      const float* __restrict__ tok_p, float* __restrict__ moe) {
    const long i = (long)blockIdx.x * 256 + threadIdx.x;   // TOK*D_MODEL
    const long t = i >> 10;
    const int n = (int)(i & 1023);
    const int s0 = tok_slot[2 * t], s1 = tok_slot[2 * t + 1];
    const float p0 = tok_p[2 * t], p1 = tok_p[2 * t + 1];
    moe[i] = p0 * y[(long)s0 * D_MODEL + n] + p1 * y[(long)s1 * D_MODEL + n];
}

// ================= host-side launch =================
extern "C" void kernel_launch(void* const* d_in, const int* in_sizes, int n_in,
                              void* d_out, int out_size, void* d_ws, size_t ws_size,
                              hipStream_t stream) {
    const float* src   = (const float*)d_in[0];
    const float* inw   = (const float*)d_in[1];
    const float* inb   = (const float*)d_in[2];
    const float* outw  = (const float*)d_in[3];
    const float* outb  = (const float*)d_in[4];
    const float* ln1g  = (const float*)d_in[5];
    const float* ln1b  = (const float*)d_in[6];
    const float* gw    = (const float*)d_in[7];
    const float* gb    = (const float*)d_in[8];
    const float* w1    = (const float*)d_in[9];
    const float* b1    = (const float*)d_in[10];
    const float* w2    = (const float*)d_in[11];
    const float* b2    = (const float*)d_in[12];
    const float* ln2g  = (const float*)d_in[13];
    const float* ln2b  = (const float*)d_in[14];

    float* out_main = (float*)d_out;                              // [4,2048,1024]
    float* out_gate = (float*)d_out + (long)TOK * D_MODEL;        // [4,2048,8]

    // ---- workspace carve-out ----
    char* ws = (char*)d_ws;
    size_t off = 0;
    auto alloc = [&](size_t bytes) -> char* {
        char* p = ws + off;
        off = (off + bytes + 255) & ~(size_t)255;
        return p;
    };
    __bf16* src_bf = (__bf16*)alloc((size_t)TOK * D_MODEL * 2);
    __bf16* inw_bf = (__bf16*)alloc((size_t)3 * D_MODEL * D_MODEL * 2);
    __bf16* outw_bf= (__bf16*)alloc((size_t)D_MODEL * D_MODEL * 2);
    __bf16* w1_bf  = (__bf16*)alloc((size_t)NEXP * D_HID * D_MODEL * 2);
    __bf16* w2_bf  = (__bf16*)alloc((size_t)NEXP * D_MODEL * D_HID * 2);
    __bf16* qb     = (__bf16*)alloc((size_t)TOK * D_MODEL * 2);
    __bf16* kb     = (__bf16*)alloc((size_t)TOK * D_MODEL * 2);
    __bf16* vt     = (__bf16*)alloc((size_t)TOK * D_MODEL * 2);
    __bf16* ctx    = (__bf16*)alloc((size_t)TOK * D_MODEL * 2);
    float*  resid1 = (float*) alloc((size_t)TOK * D_MODEL * 4);
    float*  xf     = (float*) alloc((size_t)TOK * D_MODEL * 4);
    __bf16* xbf    = (__bf16*)alloc((size_t)TOK * D_MODEL * 2);
    float*  moe    = (float*) alloc((size_t)TOK * D_MODEL * 4);
    __bf16* hbuf   = (__bf16*)alloc((size_t)TOK * D_HID * 2);
    float*  ybuf   = (float*) alloc((size_t)2 * TOK * D_MODEL * 4);
    int*    counts = (int*)   alloc(8 * 4);
    int*    offs   = (int*)   alloc(8 * 4);
    int*    tok_e  = (int*)   alloc((size_t)2 * TOK * 4);
    int*    tok_pos= (int*)   alloc((size_t)2 * TOK * 4);
    float*  tok_p  = (float*) alloc((size_t)2 * TOK * 4);
    int*    token_of=(int*)   alloc((size_t)2 * TOK * 4);
    int*    tok_slot=(int*)   alloc((size_t)2 * TOK * 4);
    if (off > ws_size) return;   // workspace too small; nothing safe to do

    auto cvt = [&](const float* s, __bf16* d, long n) {
        k_cvt<<<(unsigned)((n + 255) / 256), 256, 0, stream>>>(s, d, n);
    };
    // fp32 -> bf16 staging
    cvt(src,  src_bf, (long)TOK * D_MODEL);
    cvt(inw,  inw_bf, (long)3 * D_MODEL * D_MODEL);
    cvt(outw, outw_bf,(long)D_MODEL * D_MODEL);
    cvt(w1,   w1_bf,  (long)NEXP * D_HID * D_MODEL);
    cvt(w2,   w2_bf,  (long)NEXP * D_MODEL * D_HID);

    // QKV projection
    k_qkv<<<dim3(3 * D_MODEL / 64, TOK / 32), 32, 0, stream>>>(src_bf, inw_bf, inb,
                                                               qb, kb, vt);
    // flash attention
    k_attn<<<dim3(SEQ / 16, BATCH * NHEAD), 32, 0, stream>>>(qb, kb, vt, ctx);
    // out-proj + residual
    k_outproj<<<dim3(D_MODEL / 64, TOK / 32), 32, 0, stream>>>(ctx, outw_bf, outb,
                                                               src, resid1);
    // LN1 -> x (fp32 + bf16)
    k_ln<<<TOK, 256, 0, stream>>>(resid1, (const float*)nullptr, ln1g, ln1b, xf, xbf);
    // gate logits (second output section)
    k_gate<<<(TOK * NEXP) / 256, 256, 0, stream>>>(xf, gw, gb, out_gate);
    // top-2 routing
    k_zero8<<<1, 32, 0, stream>>>(counts);
    k_route1<<<TOK / 256, 256, 0, stream>>>(out_gate, counts, tok_e, tok_pos, tok_p);
    k_prefix<<<1, 32, 0, stream>>>(counts, offs);
    k_route2<<<(2 * TOK) / 256, 256, 0, stream>>>(offs, tok_e, tok_pos, token_of, tok_slot);
    // experts (sequential on stream; hbuf reused per expert)
    for (int e = 0; e < NEXP; ++e) {
        k_ffn1<<<dim3(D_HID / 64, TOK / 32), 32, 0, stream>>>(xbf, w1_bf, b1, counts,
                                                              offs, token_of, hbuf, e);
        k_ffn2<<<dim3(D_MODEL / 64, TOK / 32), 32, 0, stream>>>(hbuf, w2_bf, b2, counts,
                                                                offs, ybuf, e);
    }
    // combine + LN2 -> main output
    k_moe<<<(unsigned)(((long)TOK * D_MODEL) / 256), 256, 0, stream>>>(ybuf, tok_slot,
                                                                       tok_p, moe);
    k_ln<<<TOK, 256, 0, stream>>>(xf, moe, ln2g, ln2b, out_main, (__bf16*)nullptr);
}
